// Physics_Attention_Structured_Mesh_3D_49709951484187
// MI455X (gfx1250) — matrix operand order repaired and checked by reading, not compiled, not run on hardware
//
#include <hip/hip_runtime.h>

// ---------------------------------------------------------------------------
// Physics_Attention_Structured_Mesh_3D for MI455X (gfx1250, wave32, WMMA)
//
// All heavy math routed through v_wmma_f32_16x16x32_bf16 (f32 accumulate):
//  - 3x3x3 conv (64->256) as implicit GEMM, 27 taps x K=64, x resident in L2
//  - slice logits (K=32 = one WMMA step)
//  - slice-token gather  (w^T x fx) via LDS transpose + WMMA
//  - slice scatter + final 256->64 GEMM fused in one kernel
// Deterministic: partial-sum buffers per range, no float atomics.
// Workspace requirement: ~313 MB.
//
// Fragment layouts per ISA 7.12.2 / 7.12.4:
//  A (16-bit 16x32): lane = row M; lanes 0-15 hold K {ko..ko+7, ko+16..ko+23},
//                    lanes 16-31 hold K {ko+8..ko+15, ko+24..ko+31}.
//  B (16-bit 32x16): lane = col N; lanes 0-15 hold K {ko..ko+7, ko+16..ko+23},
//                    lanes 16-31 hold K {ko+8..ko+15, ko+24..ko+31}.
//  C/D (f32 16x16):  lane = col N; VGPR v holds row v (lanes 0-15) / v+8.
// ---------------------------------------------------------------------------

#define DEV __device__ __forceinline__

typedef __attribute__((ext_vector_type(16))) __bf16 bf16x16;
typedef __attribute__((ext_vector_type(8)))  float  f32x8;
typedef __attribute__((ext_vector_type(4)))  float  f32x4;

constexpr int B_     = 4;
constexpr int N_     = 65536;
constexpr int DIM_   = 64;
constexpr int HEADS_ = 8;
constexpr int DH_    = 32;
constexpr int INNER_ = 256;
constexpr int G_     = 64;     // SLICES
constexpr int GD_ = 32, GH_ = 32, GW_ = 32;
constexpr int NB_ = GD_ * GH_ * GW_;          // 32768
constexpr int RANGES_ = 32;
constexpr int RTOK_   = N_ / RANGES_;         // 2048
constexpr int CHUNK_  = 128;
constexpr int NCHUNK_ = RTOK_ / CHUNK_;       // 16

struct Bits256 { f32x4 lo, hi; };

DEV unsigned short f32_to_bf16(float f) {
  unsigned int u = __builtin_bit_cast(unsigned int, f);
  u += 0x7fffu + ((u >> 16) & 1u);
  return (unsigned short)(u >> 16);
}

DEV bf16x16 frag_zero() {
  Bits256 z;
  z.lo = f32x4{0.f, 0.f, 0.f, 0.f};
  z.hi = f32x4{0.f, 0.f, 0.f, 0.f};
  return __builtin_bit_cast(bf16x16, z);
}

// A-operand fragment: row points at this lane's M-row (16B aligned; row
// length >= ko+32 halves).  Lane-half interleave per ISA A-matrix table.
DEV bf16x16 load_a_frag(const unsigned short* row, int ko, unsigned lane) {
  const int base = ko + ((lane & 16u) ? 8 : 0);
  Bits256 r;
  r.lo = *(const f32x4*)(row + base);
  r.hi = *(const f32x4*)(row + base + 16);
  return __builtin_bit_cast(bf16x16, r);
}

// B-operand fragment: row points at this lane's N-column stored as a
// contiguous K-major row (B^T row-major).  Same K interleave as A:
// lanes 0-15 take K={ko..ko+7, ko+16..ko+23}, lanes 16-31 take K={ko+8..ko+15, ko+24..ko+31}.
DEV bf16x16 load_b_frag(const unsigned short* row, int ko, unsigned lane) {
  const int base = ko + ((lane & 16u) ? 8 : 0);
  Bits256 r;
  r.lo = *(const f32x4*)(row + base);
  r.hi = *(const f32x4*)(row + base + 16);
  return __builtin_bit_cast(bf16x16, r);
}

DEV f32x8 wmma_bf16(bf16x16 a, bf16x16 b, f32x8 c) {
  return __builtin_amdgcn_wmma_f32_16x16x32_bf16(
      /*neg_a=*/false, a, /*neg_b=*/false, b,
      /*c_mod=*/(short)0, c, /*reuse_a=*/false, /*reuse_b=*/false);
}

DEV float half16_max(float v) {
#pragma unroll
  for (int m = 1; m < 16; m <<= 1) v = fmaxf(v, __shfl_xor(v, m, 32));
  return v;
}
DEV float half16_sum(float v) {
#pragma unroll
  for (int m = 1; m < 16; m <<= 1) v += __shfl_xor(v, m, 32);
  return v;
}

// ---------------------------------------------------------------------------
// Kernel 1: x (f32) -> bf16, vectorized 4-wide
// ---------------------------------------------------------------------------
__global__ void convert_x_kernel(const float* __restrict__ x,
                                 unsigned short* __restrict__ xbf, int n4) {
  int i = blockIdx.x * blockDim.x + threadIdx.x;
  for (; i < n4; i += gridDim.x * blockDim.x) {
    f32x4 v = ((const f32x4*)x)[i];
    unsigned long long pk =
        (unsigned long long)f32_to_bf16(v.x) |
        ((unsigned long long)f32_to_bf16(v.y) << 16) |
        ((unsigned long long)f32_to_bf16(v.z) << 32) |
        ((unsigned long long)f32_to_bf16(v.w) << 48);
    ((unsigned long long*)xbf)[i] = pk;
  }
}

// ---------------------------------------------------------------------------
// Kernel 2: repack + convert all weights to bf16.
//   conv (O,I,3,3,3) -> wtap[tap][o][i]   (B-operand rows = o, K = i)
//   lin (O,I) -> as-is; slice_w (G,DH) -> as-is; out_w (DIM,INNER) -> as-is
// ---------------------------------------------------------------------------
constexpr int WTAP_ELTS = 27 * INNER_ * DIM_;   // 442368
constexpr int LIN_ELTS  = INNER_ * DIM_;        // 16384
constexpr int SLW_ELTS  = G_ * DH_;             // 2048
constexpr int OUTW_ELTS = DIM_ * INNER_;        // 16384
constexpr int PREP_TOT  = 2 * WTAP_ELTS + 2 * LIN_ELTS + SLW_ELTS + OUTW_ELTS;

__global__ void prep_weights_kernel(
    const float* __restrict__ fxcw, const float* __restrict__ xpcw,
    const float* __restrict__ fxlw, const float* __restrict__ xplw,
    const float* __restrict__ slw,  const float* __restrict__ outw,
    unsigned short* __restrict__ wtap_fx, unsigned short* __restrict__ wtap_xp,
    unsigned short* __restrict__ lin_fx,  unsigned short* __restrict__ lin_xp,
    unsigned short* __restrict__ slw_bf,  unsigned short* __restrict__ outw_bf) {
  int i = blockIdx.x * blockDim.x + threadIdx.x;
  for (; i < PREP_TOT; i += gridDim.x * blockDim.x) {
    int j = i;
    if (j < 2 * WTAP_ELTS) {
      const float* src = (j < WTAP_ELTS) ? fxcw : xpcw;
      unsigned short* dst = (j < WTAP_ELTS) ? wtap_fx : wtap_xp;
      const int k = j % WTAP_ELTS;
      const int ii = k & 63, o = (k >> 6) & 255, tap = k >> 14;
      dst[k] = f32_to_bf16(src[(o * 64 + ii) * 27 + tap]);
      continue;
    }
    j -= 2 * WTAP_ELTS;
    if (j < 2 * LIN_ELTS) {
      if (j < LIN_ELTS) lin_fx[j] = f32_to_bf16(fxlw[j]);
      else              lin_xp[j - LIN_ELTS] = f32_to_bf16(xplw[j - LIN_ELTS]);
      continue;
    }
    j -= 2 * LIN_ELTS;
    if (j < SLW_ELTS) { slw_bf[j] = f32_to_bf16(slw[j]); continue; }
    j -= SLW_ELTS;
    outw_bf[j] = f32_to_bf16(outw[j]);
  }
}

// ---------------------------------------------------------------------------
// Kernel 3: projection (conv region as implicit GEMM + linear region)
//   grid: B * 2048 blocks; block 256 = 8 waves.
//   wave w: 16 tokens (msub=w>>2), 64 outputs (ogrp=w&3) -> 4 WMMA tiles.
// ---------------------------------------------------------------------------
__global__ __launch_bounds__(256)
void proj_kernel(const unsigned short* __restrict__ xbf,
                 const unsigned short* __restrict__ wtap,
                 const float* __restrict__ conv_b,
                 const unsigned short* __restrict__ wlin,
                 const float* __restrict__ lin_b,
                 unsigned short* __restrict__ mid) {
  const unsigned tid = threadIdx.x;
  const unsigned lane = tid & 31u;
  const unsigned wv = tid >> 5;
  const int msub = (int)(wv >> 2);
  const int obase = (int)(wv & 3u) * 64;
  const int r = blockIdx.x & 2047;
  const int b = blockIdx.x >> 11;

  const f32x8 fz = {0.f, 0.f, 0.f, 0.f, 0.f, 0.f, 0.f, 0.f};
  f32x8 acc[4] = {fz, fz, fz, fz};
  int tok16;
  const float* bias;

  if (r < 1024) {  // structured (conv) region: d=r>>5, h=r&31, w in [0,32)
    const int d = r >> 5, h = r & 31;
    tok16 = (d * GH_ + h) * GW_ + msub * 16;
    bias = conv_b;
    for (int dz = -1; dz <= 1; ++dz) {
      const int dd = d + dz;
      if ((unsigned)dd >= (unsigned)GD_) continue;   // zero padding
      for (int dy = -1; dy <= 1; ++dy) {
        const int hh = h + dy;
        if ((unsigned)hh >= (unsigned)GH_) continue;
        for (int dx = -1; dx <= 1; ++dx) {
          const int tap = (dz + 1) * 9 + (dy + 1) * 3 + (dx + 1);
          const int wt = msub * 16 + (int)(lane & 15u) + dx;
          const bool valid = (unsigned)wt < (unsigned)GW_;
          const unsigned short* arow =
              xbf + ((size_t)(b * N_) + (size_t)((dd * GH_ + hh) * GW_ + wt)) * DIM_;
          for (int ko = 0; ko < DIM_; ko += 32) {
            const bf16x16 a = valid ? load_a_frag(arow, ko, lane) : frag_zero();
#pragma unroll
            for (int f = 0; f < 4; ++f) {
              const int o = obase + f * 16 + (int)(lane & 15u);
              const unsigned short* brow = wtap + ((size_t)tap * INNER_ + o) * DIM_;
              acc[f] = wmma_bf16(a, load_b_frag(brow, ko, lane), acc[f]);
            }
          }
        }
      }
    }
  } else {  // unstructured (linear) region
    tok16 = NB_ + (r - 1024) * 32 + msub * 16;
    bias = lin_b;
    const int trow = tok16 + (int)(lane & 15u);
    const unsigned short* arow = xbf + ((size_t)(b * N_) + trow) * DIM_;
    for (int ko = 0; ko < DIM_; ko += 32) {
      const bf16x16 a = load_a_frag(arow, ko, lane);
#pragma unroll
      for (int f = 0; f < 4; ++f) {
        const int o = obase + f * 16 + (int)(lane & 15u);
        acc[f] = wmma_bf16(a, load_b_frag(wlin + (size_t)o * DIM_, ko, lane), acc[f]);
      }
    }
  }

  const int mh = (lane & 16u) ? 8 : 0;
#pragma unroll
  for (int f = 0; f < 4; ++f) {
    const int o = obase + f * 16 + (int)(lane & 15u);
    const float bo = bias[o];
#pragma unroll
    for (int v = 0; v < 8; ++v) {
      const int token = tok16 + v + mh;
      mid[((size_t)(b * N_) + token) * INNER_ + o] = f32_to_bf16(acc[f][v] + bo);
    }
  }
}

// ---------------------------------------------------------------------------
// Kernel 4: per (b,h,range): logits -> softmax -> partial slice_token & norm.
//   phase 1: WMMA logits (K=32), wave softmax, transpose weights into LDS
//   phase 2: WMMA  wT[64g x 128n] x fxT[128n x 32c] accumulated over chunks
// ---------------------------------------------------------------------------
__global__ __launch_bounds__(256)
void slice_reduce_kernel(const unsigned short* __restrict__ xmid,
                         const unsigned short* __restrict__ fxmid,
                         const unsigned short* __restrict__ slw_bf,
                         const float* __restrict__ slb,
                         const float* __restrict__ temp,
                         float* __restrict__ part_tok,    // [B,H,RANGES,64,32]
                         float* __restrict__ part_norm) { // [B,H,RANGES,64]
  __shared__ unsigned short sWT[64 * 136];   // w^T  [g][n] bf16
  __shared__ unsigned short sFX[32 * 136];   // fx^T [c][n] bf16
  __shared__ float sN[8][64];                // per-wave norm partials

  const unsigned tid = threadIdx.x, lane = tid & 31u, wv = tid >> 5;
  const int rng = blockIdx.x & (RANGES_ - 1);
  const int h = (blockIdx.x >> 5) & 7;
  const int b = blockIdx.x >> 8;
  const int tok0 = rng * RTOK_;

  const float t = fminf(fmaxf(temp[h], 0.1f), 5.0f);
  const float it = 1.0f / t;

  for (int i = tid; i < 8 * 64; i += 256) ((float*)sN)[i] = 0.f;
  __syncthreads();

  const f32x8 fz = {0.f, 0.f, 0.f, 0.f, 0.f, 0.f, 0.f, 0.f};
  f32x8 acc = fz;                       // this wave's [16g x 16c] tile
  const int mt = (int)(wv >> 1), nt = (int)(wv & 1u);
  const int mh = (lane & 16u) ? 8 : 0;

  for (int ch = 0; ch < NCHUNK_; ++ch) {
    const int cb = tok0 + ch * CHUNK_;
    // stage fx chunk transposed (u64 loads, 4 LDS scatters): sFX[c][n]
    for (int i = tid; i < CHUNK_ * DH_ / 4; i += 256) {
      const int n = i >> 3;
      const int c4 = (i & 7) * 4;
      const unsigned long long q = *(const unsigned long long*)(
          fxmid + ((size_t)(b * N_) + cb + n) * INNER_ + h * DH_ + c4);
      sFX[(c4 + 0) * 136 + n] = (unsigned short)(q);
      sFX[(c4 + 1) * 136 + n] = (unsigned short)(q >> 16);
      sFX[(c4 + 2) * 136 + n] = (unsigned short)(q >> 32);
      sFX[(c4 + 3) * 136 + n] = (unsigned short)(q >> 48);
    }
    // phase 1: logits + softmax for this wave's 16 tokens
    {
      const int trow = cb + (int)wv * 16 + (int)(lane & 15u);
      const unsigned short* arow = xmid + ((size_t)(b * N_) + trow) * INNER_ + h * DH_;
      const bf16x16 a = load_a_frag(arow, 0, lane);
      f32x8 lg[4];
#pragma unroll
      for (int f = 0; f < 4; ++f) {
        const int g = f * 16 + (int)(lane & 15u);
        lg[f] = wmma_bf16(a, load_b_frag(slw_bf + (size_t)g * DH_, 0, lane), fz);
      }
#pragma unroll
      for (int f = 0; f < 4; ++f) {
        const float bg = slb[f * 16 + (lane & 15u)];
#pragma unroll
        for (int v = 0; v < 8; ++v) lg[f][v] = (lg[f][v] + bg) * it;
      }
#pragma unroll
      for (int v = 0; v < 8; ++v) {
        float mx = fmaxf(fmaxf(lg[0][v], lg[1][v]), fmaxf(lg[2][v], lg[3][v]));
        mx = half16_max(mx);
        float e[4], s = 0.f;
#pragma unroll
        for (int f = 0; f < 4; ++f) { e[f] = __expf(lg[f][v] - mx); s += e[f]; }
        s = half16_sum(s);
        const float inv = 1.0f / s;
        const int nloc = (int)wv * 16 + v + mh;
#pragma unroll
        for (int f = 0; f < 4; ++f) {
          const float w = e[f] * inv;
          lg[f][v] = w;
          sWT[(f * 16 + (lane & 15u)) * 136 + nloc] = f32_to_bf16(w);
        }
      }
#pragma unroll
      for (int f = 0; f < 4; ++f) {
        float s = 0.f;
#pragma unroll
        for (int v = 0; v < 8; ++v) s += lg[f][v];
        s += __shfl_xor(s, 16, 32);
        if (lane < 16) sN[wv][f * 16 + lane] += s;
      }
    }
    __syncthreads();
    // phase 2: acc += wT x fxT, K = 128
#pragma unroll
    for (int ks = 0; ks < 4; ++ks) {
      const int g = mt * 16 + (int)(lane & 15u);
      const bf16x16 a = load_a_frag(sWT + g * 136, ks * 32, lane);
      const int c = nt * 16 + (int)(lane & 15u);
      const bf16x16 bb = load_b_frag(sFX + c * 136, ks * 32, lane);
      acc = wmma_bf16(a, bb, acc);
    }
    __syncthreads();
  }

  {
    float* dst = part_tok + ((size_t)(b * HEADS_ + h) * RANGES_ + rng) * (G_ * DH_);
#pragma unroll
    for (int v = 0; v < 8; ++v) {
      const int g = mt * 16 + v + mh;
      const int c = nt * 16 + (int)(lane & 15u);
      dst[g * DH_ + c] = acc[v];
    }
  }
  if (tid < 64) {
    float s = 0.f;
#pragma unroll
    for (int w = 0; w < 8; ++w) s += sN[w][tid];
    part_norm[((size_t)(b * HEADS_ + h) * RANGES_ + rng) * G_ + tid] = s;
  }
}

// ---------------------------------------------------------------------------
// Kernel 5: reduce partials, q/k/v, 64x64 attention, emit out_slice^T (bf16)
//   32 blocks (b,h).  Negligible FLOPs -> plain VALU.
// ---------------------------------------------------------------------------
__global__ __launch_bounds__(256)
void attn_kernel(const float* __restrict__ part_tok,
                 const float* __restrict__ part_norm,
                 const float* __restrict__ wq, const float* __restrict__ wk,
                 const float* __restrict__ wv_,
                 unsigned short* __restrict__ oslT) {  // [B*H][32c][64g]
  __shared__ float sST[G_ * DH_];
  __shared__ float sQ[G_ * DH_], sK[G_ * DH_], sV[G_ * DH_];
  __shared__ float sNrm[G_];
  const int tid = threadIdx.x;
  const int bh = blockIdx.x;
  const float* pt = part_tok + (size_t)bh * RANGES_ * G_ * DH_;
  const float* pn = part_norm + (size_t)bh * RANGES_ * G_;

  if (tid < G_) {
    float s = 0.f;
    for (int r = 0; r < RANGES_; ++r) s += pn[r * G_ + tid];
    sNrm[tid] = s + 1e-5f;
  }
  __syncthreads();
  for (int i = tid; i < G_ * DH_; i += 256) {
    float s = 0.f;
    for (int r = 0; r < RANGES_; ++r) s += pt[r * G_ * DH_ + i];
    sST[i] = s / sNrm[i >> 5];
  }
  __syncthreads();
  for (int i = tid; i < G_ * DH_; i += 256) {
    const int g = i >> 5, c = i & 31;
    float q = 0.f, k = 0.f, v = 0.f;
    const float* srow = sST + g * DH_;
    for (int kk = 0; kk < DH_; ++kk) {
      const float xv = srow[kk];
      q += xv * wq[c * DH_ + kk];
      k += xv * wk[c * DH_ + kk];
      v += xv * wv_[c * DH_ + kk];
    }
    sQ[i] = q; sK[i] = k; sV[i] = v;
  }
  __syncthreads();
  if (tid < G_) {
    const int g = tid;
    const float scale = 0.17677669529663687f;  // 32^-0.5
    float dots[G_];
    float mx = -1e30f;
    for (int j = 0; j < G_; ++j) {
      float d = 0.f;
      for (int c = 0; c < DH_; ++c) d += sQ[g * DH_ + c] * sK[j * DH_ + c];
      d *= scale; dots[j] = d; mx = fmaxf(mx, d);
    }
    float s = 0.f;
    for (int j = 0; j < G_; ++j) { dots[j] = __expf(dots[j] - mx); s += dots[j]; }
    const float inv = 1.f / s;
    float o[DH_];
    for (int c = 0; c < DH_; ++c) o[c] = 0.f;
    for (int j = 0; j < G_; ++j) {
      const float p = dots[j] * inv;
      for (int c = 0; c < DH_; ++c) o[c] += p * sV[j * DH_ + c];
    }
    for (int c = 0; c < DH_; ++c)
      oslT[(size_t)bh * DH_ * G_ + c * G_ + g] = f32_to_bf16(o[c]);
  }
}

// ---------------------------------------------------------------------------
// Kernel 6: fused scatter + output GEMM.
//   Per 16-token tile, per head: recompute slice weights (WMMA logits +
//   softmax), WMMA  w[16n x 64g] x out_slice^T -> out_x columns h*32..,
//   then fused WMMA GEMM (K=256) against out_w + bias -> f32 output.
//   Block = 128 threads (4 independent waves), per-wave LDS buffers.
// ---------------------------------------------------------------------------
__global__ __launch_bounds__(128)
void output_kernel(const unsigned short* __restrict__ xmid,
                   const unsigned short* __restrict__ slw_bf,
                   const float* __restrict__ slb,
                   const float* __restrict__ temp,
                   const unsigned short* __restrict__ oslT,   // [B*H][32][64]
                   const unsigned short* __restrict__ outw_bf,// [64][256]
                   const float* __restrict__ outb,
                   float* __restrict__ out) {                 // [B,N,64]
  __shared__ unsigned short sW[4 * 16 * 72];    // per-wave w   [n][g]
  __shared__ unsigned short sO[4 * 16 * 264];   // per-wave out_x [n][256]
  const unsigned tid = threadIdx.x, lane = tid & 31u, wv = tid >> 5;
  const int b = blockIdx.x >> 10;               // N/64 = 1024 blocks per b
  const int cb = (blockIdx.x & 1023) * 64;
  const int tb = cb + (int)wv * 16;
  unsigned short* wbuf = sW + wv * (16 * 72);
  unsigned short* obuf = sO + wv * (16 * 264);
  const int mh = (lane & 16u) ? 8 : 0;
  const f32x8 fz = {0.f, 0.f, 0.f, 0.f, 0.f, 0.f, 0.f, 0.f};

  for (int h = 0; h < HEADS_; ++h) {
    const float t = fminf(fmaxf(temp[h], 0.1f), 5.0f);
    const float it = 1.0f / t;
    // logits + softmax -> wbuf[n][g]
    const int trow = tb + (int)(lane & 15u);
    const unsigned short* arow = xmid + ((size_t)(b * N_) + trow) * INNER_ + h * DH_;
    const bf16x16 a = load_a_frag(arow, 0, lane);
    f32x8 lg[4];
#pragma unroll
    for (int f = 0; f < 4; ++f) {
      const int g = f * 16 + (int)(lane & 15u);
      lg[f] = wmma_bf16(a, load_b_frag(slw_bf + (size_t)g * DH_, 0, lane), fz);
    }
#pragma unroll
    for (int f = 0; f < 4; ++f) {
      const float bg = slb[f * 16 + (lane & 15u)];
#pragma unroll
      for (int v = 0; v < 8; ++v) lg[f][v] = (lg[f][v] + bg) * it;
    }
#pragma unroll
    for (int v = 0; v < 8; ++v) {
      float mx = fmaxf(fmaxf(lg[0][v], lg[1][v]), fmaxf(lg[2][v], lg[3][v]));
      mx = half16_max(mx);
      float e[4], s = 0.f;
#pragma unroll
      for (int f = 0; f < 4; ++f) { e[f] = __expf(lg[f][v] - mx); s += e[f]; }
      s = half16_sum(s);
      const float inv = 1.0f / s;
      const int row = v + mh;
#pragma unroll
      for (int f = 0; f < 4; ++f)
        wbuf[row * 72 + f * 16 + (lane & 15u)] = f32_to_bf16(e[f] * inv);
    }
    // out_head[16n x 32c] = w x out_slice^T   (K = 64 over g)
    f32x8 oc[2] = {fz, fz};
    const unsigned short* bT = oslT + (size_t)(b * HEADS_ + h) * DH_ * G_;
    for (int ko = 0; ko < G_; ko += 32) {
      const bf16x16 aw = load_a_frag(wbuf + (lane & 15u) * 72, ko, lane);
#pragma unroll
      for (int nc = 0; nc < 2; ++nc) {
        const int c = nc * 16 + (int)(lane & 15u);
        oc[nc] = wmma_bf16(aw, load_b_frag(bT + c * G_, ko, lane), oc[nc]);
      }
    }
#pragma unroll
    for (int nc = 0; nc < 2; ++nc) {
      const int col = h * DH_ + nc * 16 + (int)(lane & 15u);
#pragma unroll
      for (int v = 0; v < 8; ++v)
        obuf[(v + mh) * 264 + col] = f32_to_bf16(oc[nc][v]);
    }
  }
  // final: [16n x 64dim] = obuf[16n x 256] @ out_w^T  (K = 256)
  f32x8 fo[4] = {fz, fz, fz, fz};
  for (int ko = 0; ko < INNER_; ko += 32) {
    const bf16x16 a = load_a_frag(obuf + (lane & 15u) * 264, ko, lane);
#pragma unroll
    for (int f = 0; f < 4; ++f) {
      const int dimc = f * 16 + (int)(lane & 15u);
      fo[f] = wmma_bf16(a, load_b_frag(outw_bf + (size_t)dimc * INNER_, ko, lane), fo[f]);
    }
  }
#pragma unroll
  for (int f = 0; f < 4; ++f) {
    const int dimc = f * 16 + (int)(lane & 15u);
    const float bb = outb[dimc];
#pragma unroll
    for (int v = 0; v < 8; ++v) {
      const int token = tb + v + mh;
      out[((size_t)(b * N_) + token) * DIM_ + dimc] = fo[f][v] + bb;
    }
  }
}

// ---------------------------------------------------------------------------
extern "C" void kernel_launch(void* const* d_in, const int* in_sizes, int n_in,
                              void* d_out, int out_size, void* d_ws, size_t ws_size,
                              hipStream_t stream) {
  (void)in_sizes; (void)n_in; (void)out_size; (void)ws_size;
  const float* x    = (const float*)d_in[0];
  const float* temp = (const float*)d_in[1];
  const float* fxcw = (const float*)d_in[2];
  const float* fxcb = (const float*)d_in[3];
  const float* fxlw = (const float*)d_in[4];
  const float* fxlb = (const float*)d_in[5];
  const float* xpcw = (const float*)d_in[6];
  const float* xpcb = (const float*)d_in[7];
  const float* xplw = (const float*)d_in[8];
  const float* xplb = (const float*)d_in[9];
  const float* slw  = (const float*)d_in[10];
  const float* slb  = (const float*)d_in[11];
  const float* wq   = (const float*)d_in[12];
  const float* wk   = (const float*)d_in[13];
  const float* wvv  = (const float*)d_in[14];
  const float* outw = (const float*)d_in[15];
  const float* outb = (const float*)d_in[16];
  float* out = (float*)d_out;

  char* ws = (char*)d_ws;
  size_t off = 0;
  auto alloc = [&](size_t bytes) -> char* {
    char* p = ws + off;
    off = (off + bytes + 255) & ~(size_t)255;
    return p;
  };
  unsigned short* xbf     = (unsigned short*)alloc((size_t)B_ * N_ * DIM_ * 2);
  unsigned short* mid_fx  = (unsigned short*)alloc((size_t)B_ * N_ * INNER_ * 2);
  unsigned short* mid_xp  = (unsigned short*)alloc((size_t)B_ * N_ * INNER_ * 2);
  unsigned short* wtap_fx = (unsigned short*)alloc((size_t)WTAP_ELTS * 2);
  unsigned short* wtap_xp = (unsigned short*)alloc((size_t)WTAP_ELTS * 2);
  unsigned short* lin_fx  = (unsigned short*)alloc((size_t)LIN_ELTS * 2);
  unsigned short* lin_xp  = (unsigned short*)alloc((size_t)LIN_ELTS * 2);
  unsigned short* slw_bf  = (unsigned short*)alloc((size_t)SLW_ELTS * 2);
  unsigned short* outw_bf = (unsigned short*)alloc((size_t)OUTW_ELTS * 2);
  float* part_tok  = (float*)alloc((size_t)B_ * HEADS_ * RANGES_ * G_ * DH_ * 4);
  float* part_norm = (float*)alloc((size_t)B_ * HEADS_ * RANGES_ * G_ * 4);
  unsigned short* oslT = (unsigned short*)alloc((size_t)B_ * HEADS_ * DH_ * G_ * 2);

  convert_x_kernel<<<4096, 256, 0, stream>>>(x, xbf, B_ * N_ * DIM_ / 4);
  prep_weights_kernel<<<1024, 256, 0, stream>>>(fxcw, xpcw, fxlw, xplw, slw, outw,
                                                wtap_fx, wtap_xp, lin_fx, lin_xp,
                                                slw_bf, outw_bf);
  proj_kernel<<<B_ * 2048, 256, 0, stream>>>(xbf, wtap_fx, fxcb, lin_fx, fxlb, mid_fx);
  proj_kernel<<<B_ * 2048, 256, 0, stream>>>(xbf, wtap_xp, xpcb, lin_xp, xplb, mid_xp);
  slice_reduce_kernel<<<B_ * HEADS_ * RANGES_, 256, 0, stream>>>(
      mid_xp, mid_fx, slw_bf, slb, temp, part_tok, part_norm);
  attn_kernel<<<B_ * HEADS_, 256, 0, stream>>>(part_tok, part_norm, wq, wk, wvv, oslT);
  output_kernel<<<B_ * (N_ / 64), 128, 0, stream>>>(
      mid_xp, slw_bf, slb, temp, oslT, outw_bf, outb, out);
}
